// linear_trunc_dep_abs_3762391352097
// MI455X (gfx1250) — compile-verified, matrix operand order checked by reading, not run on hardware
//
#include <hip/hip_runtime.h>
#include <hip/hip_bf16.h>

typedef __attribute__((ext_vector_type(16))) __bf16      v16bf;
typedef __attribute__((ext_vector_type(8)))  float       v8f;
typedef __attribute__((ext_vector_type(4)))  unsigned int v4u;
typedef __attribute__((ext_vector_type(4)))  int          v4i;

#define DIM  4096           // B == IN == OUT == 4096
#define BM   128            // block tile M
#define BN   256            // block tile N
#define BK   32             // K step (WMMA K)
#define LDST 40             // LDS row stride in bf16 elems (16B-aligned, staggered banks)

#if __has_builtin(__builtin_amdgcn_global_load_async_to_lds_b128) && \
    __has_builtin(__builtin_amdgcn_s_wait_asynccnt)
#define USE_ASYNC 1
typedef v4i __attribute__((address_space(1)))* gas_v4i;   // global (prints as __device__)
typedef v4i __attribute__((address_space(3)))* las_v4i;   // LDS
#else
#define USE_ASYNC 0
#endif

__device__ __forceinline__ unsigned short f2bf(float f) {
    unsigned int u = __float_as_uint(f);
    unsigned int r = u + 0x7FFFu + ((u >> 16) & 1u);   // round-to-nearest-even
    return (unsigned short)(r >> 16);
}

// ---------------- K1: colsum[i] = sum_o W[o][i] ----------------
__global__ __launch_bounds__(256) void colsum_kernel(const float* __restrict__ W,
                                                     float* __restrict__ colsum) {
    int i = blockIdx.x * 256 + threadIdx.x;
    float s = 0.f;
    for (int o = 0; o < DIM; ++o)
        s += W[(size_t)o * DIM + i];     // coalesced across lanes
    colsum[i] = s;
}

// ---------------- K2: W (f32) -> Wb (bf16) ----------------
__global__ __launch_bounds__(256) void cvtW_kernel(const float* __restrict__ W,
                                                   unsigned short* __restrict__ Wb) {
    size_t i = ((size_t)blockIdx.x * 256 + threadIdx.x) * 4;
    float4 f = *(const float4*)(W + i);
    unsigned int lo = (unsigned int)f2bf(f.x) | ((unsigned int)f2bf(f.y) << 16);
    unsigned int hi = (unsigned int)f2bf(f.z) | ((unsigned int)f2bf(f.w) << 16);
    *(uint2*)(Wb + i) = make_uint2(lo, hi);
}

// ------- K3: per-row top-k on |x*colsum| -> masked bf16 row of x -------
__global__ __launch_bounds__(256) void topk_mask_kernel(const float* __restrict__ x,
                                                        const float* __restrict__ colsum,
                                                        const int*   __restrict__ kptr,
                                                        unsigned short* __restrict__ Xb) {
    __shared__ unsigned int keys[DIM];
    __shared__ unsigned int cnt;
    const int b = blockIdx.x;
    const unsigned int k = (unsigned int)kptr[0];
    const float* xr = x + (size_t)b * DIM;

    for (int i = threadIdx.x; i < DIM; i += 256)
        keys[i] = __float_as_uint(fabsf(xr[i] * colsum[i]));
    __syncthreads();

    unsigned int T = 0;
    for (int bit = 31; bit >= 0; --bit) {
        unsigned int cand = T | (1u << bit);
        if (threadIdx.x == 0) cnt = 0;
        __syncthreads();
        unsigned int local = 0;
        for (int i = threadIdx.x; i < DIM; i += 256)
            local += (keys[i] >= cand) ? 1u : 0u;
        atomicAdd(&cnt, local);
        __syncthreads();
        if (cnt >= k) T = cand;          // identical decision in every thread
        __syncthreads();
    }

    for (int i = threadIdx.x; i < DIM; i += 256)
        Xb[(size_t)b * DIM + i] = (keys[i] >= T) ? (unsigned short)0 : f2bf(xr[i]);
}

// ---------------- K4: out = Xb(bf16) @ Wb(bf16)^T, f32 accumulate ----------------
// Block: 256 threads = 8 waves, C tile 128x256; wave (waveM,waveN) owns 64x64
// = 16 accumulator tiles -> 16 WMMA per 16 ds_load_b128 per K step.
union Frag { v4u u[2]; v16bf bf; };

__global__ __launch_bounds__(256) void gemm_bf16_kernel(const unsigned short* __restrict__ Xb,
                                                        const unsigned short* __restrict__ Wb,
                                                        float* __restrict__ out) {
    __shared__ __align__(16) unsigned short As[2][BM * LDST];   // 2 x 10240 B
    __shared__ __align__(16) unsigned short Bs[2][BN * LDST];   // 2 x 20480 B

    const int tid   = threadIdx.x;
    const int lane  = tid & 31;
    const int wave  = tid >> 5;
    const int waveM = wave & 1;    // 2 strips of 64 rows
    const int waveN = wave >> 1;   // 4 strips of 64 cols
    const int half  = lane >> 4;   // 16-bit A layout: lane halves split K
    const int l15   = lane & 15;
    const int bm    = blockIdx.x * BM;
    const int bn    = blockIdx.y * BN;

    v8f acc[4][4];
#pragma unroll
    for (int mi = 0; mi < 4; ++mi)
#pragma unroll
        for (int ni = 0; ni < 4; ++ni)
            acc[mi][ni] = (v8f){};

    // Stage the (kk)-slab into LDS buffer `buf`.
    auto stage = [&](int kk, int buf) {
#if USE_ASYNC
        // A: 512 16B chunks, 2/thread.  B: 1024 16B chunks, 4/thread.
#pragma unroll
        for (int c = 0; c < 2; ++c) {
            int chunk = tid + c * 256;
            int row = chunk >> 2, cc = (chunk & 3) * 8;
            __builtin_amdgcn_global_load_async_to_lds_b128(
                (gas_v4i)&Xb[(size_t)(bm + row) * DIM + kk + cc],
                (las_v4i)&As[buf][row * LDST + cc], 0, 0);
        }
#pragma unroll
        for (int c = 0; c < 4; ++c) {
            int chunk = tid + c * 256;
            int row = chunk >> 2, cc = (chunk & 3) * 8;
            __builtin_amdgcn_global_load_async_to_lds_b128(
                (gas_v4i)&Wb[(size_t)(bn + row) * DIM + kk + cc],
                (las_v4i)&Bs[buf][row * LDST + cc], 0, 0);
        }
#else
#pragma unroll
        for (int c = 0; c < 2; ++c) {
            int chunk = tid + c * 256;
            int row = chunk >> 2, cc = (chunk & 3) * 8;
            *(v4u*)&As[buf][row * LDST + cc] =
                *(const v4u*)&Xb[(size_t)(bm + row) * DIM + kk + cc];
        }
#pragma unroll
        for (int c = 0; c < 4; ++c) {
            int chunk = tid + c * 256;
            int row = chunk >> 2, cc = (chunk & 3) * 8;
            *(v4u*)&Bs[buf][row * LDST + cc] =
                *(const v4u*)&Wb[(size_t)(bn + row) * DIM + kk + cc];
        }
#endif
    };

    // Consume buffer `buf`: 8 fragment loads (16 ds_load_b128), 16 WMMAs.
    auto compute = [&](int buf) {
        Frag a[4];
#pragma unroll
        for (int mi = 0; mi < 4; ++mi) {
            const unsigned short* p =
                &As[buf][(waveM * 64 + mi * 16 + l15) * LDST + half * 8];
            a[mi].u[0] = *(const v4u*)p;
            a[mi].u[1] = *(const v4u*)(p + 16);
        }
#pragma unroll
        for (int ni = 0; ni < 4; ++ni) {
            Frag bfr;
            const unsigned short* p =
                &Bs[buf][(waveN * 64 + ni * 16 + l15) * LDST + half * 8];
            bfr.u[0] = *(const v4u*)p;
            bfr.u[1] = *(const v4u*)(p + 16);
#pragma unroll
            for (int mi = 0; mi < 4; ++mi)
                acc[mi][ni] = __builtin_amdgcn_wmma_f32_16x16x32_bf16(
                    false, a[mi].bf, false, bfr.bf,
                    (short)0, acc[mi][ni], false, false);
        }
    };

    stage(0, 0);
#if USE_ASYNC
    __builtin_amdgcn_s_wait_asynccnt(0);
#endif
    __syncthreads();

    int buf = 0;
    for (int kk = 0; kk < DIM; kk += BK) {
        if (kk + BK < DIM) stage(kk + BK, buf ^ 1);
        compute(buf);
#if USE_ASYNC
        __builtin_amdgcn_s_wait_asynccnt(0);
#endif
        __syncthreads();
        buf ^= 1;
    }

    // Epilogue: f32 C/D layout — lane halves give M blocks 0..7 / 8..15 per VGPR j.
#pragma unroll
    for (int mi = 0; mi < 4; ++mi) {
        int rbase = bm + waveM * 64 + mi * 16 + half * 8;
#pragma unroll
        for (int ni = 0; ni < 4; ++ni) {
            int col = bn + waveN * 64 + ni * 16 + l15;
#pragma unroll
            for (int j = 0; j < 8; ++j)
                out[(size_t)(rbase + j) * DIM + col] = acc[mi][ni][j];
        }
    }
}

extern "C" void kernel_launch(void* const* d_in, const int* in_sizes, int n_in,
                              void* d_out, int out_size, void* d_ws, size_t ws_size,
                              hipStream_t stream) {
    const float* x    = (const float*)d_in[0];
    const float* W    = (const float*)d_in[1];
    const int*   kptr = (const int*)d_in[2];
    float*       out  = (float*)d_out;

    // Workspace layout: Wb bf16 (32MB) | Xb bf16 (32MB) | colsum f32 (16KB)
    unsigned short* Wb = (unsigned short*)d_ws;
    unsigned short* Xb = Wb + (size_t)DIM * DIM;
    float* colsum = (float*)((char*)d_ws + (size_t)2 * DIM * DIM * sizeof(unsigned short));

    colsum_kernel<<<DIM / 256, 256, 0, stream>>>(W, colsum);
    cvtW_kernel<<<(DIM * (DIM / 4)) / 256, 256, 0, stream>>>(W, Wb);
    topk_mask_kernel<<<DIM, 256, 0, stream>>>(x, colsum, kptr, Xb);

    dim3 grid(DIM / BM, DIM / BN);
    gemm_bf16_kernel<<<grid, 256, 0, stream>>>(Xb, Wb, out);
}
